// LGGCN_83305185673744
// MI455X (gfx1250) — matrix operand order repaired and checked by reading, not compiled, not run on hardware
//
#include <hip/hip_runtime.h>
#include <stdint.h>

#define BB 32
#define SS 2048
#define DD 160

typedef __attribute__((ext_vector_type(16))) __bf16 v16bf;
typedef __attribute__((ext_vector_type(8)))  float  v8f;

union Frag {
  v16bf v;
  uint32_t u[8];
  uint4 q4[2];
};

// round-to-nearest-even f32 -> bf16
__device__ __forceinline__ uint16_t bf16_1(float a){
  uint32_t u = __float_as_uint(a);
  u += 0x7FFFu + ((u >> 16) & 1u);
  return (uint16_t)(u >> 16);
}
__device__ __forceinline__ uint32_t pk_bf16(float a, float b){
  uint32_t ua = __float_as_uint(a); ua += 0x7FFFu + ((ua >> 16) & 1u);
  uint32_t ub = __float_as_uint(b); ub += 0x7FFFu + ((ub >> 16) & 1u);
  return (ua >> 16) | (ub & 0xFFFF0000u);
}

// ---- convert the three DxD weight matrices to bf16 (contiguous in ws) ----
__global__ void wcvt_kernel(const float* __restrict__ wq,
                            const float* __restrict__ wk,
                            const float* __restrict__ wv,
                            uint16_t* __restrict__ out){
  int i = blockIdx.x * blockDim.x + threadIdx.x;
  const int n = DD * DD;
  if (i < n)          out[i] = bf16_1(wq[i]);
  else if (i < 2*n)   out[i] = bf16_1(wk[i - n]);
  else if (i < 3*n)   out[i] = bf16_1(wv[i - 2*n]);
}

// ---- projection: out[m,e] = sum_d X[m,d]*W[e,d] + bias[e] ----
template<bool TRANS>
__global__ void __launch_bounds__(256, 1)
proj_kernel(const float* __restrict__ X,      // [B,S,D] f32
            const uint16_t* __restrict__ Wb,  // [D,D] bf16
            const float* __restrict__ bias,   // [D]
            uint16_t* __restrict__ out)       // [B,S,D] or [B,D,S]
{
  const int lane = threadIdx.x & 31;
  const int wave = threadIdx.x >> 5;
  const int tile = blockIdx.x * 8 + wave;
  const int tilesPerB = SS / 16;
  if (tile >= BB * tilesPerB) return;
  const int b   = tile / tilesPerB;
  const int m0  = (tile % tilesPerB) * 16;
  const int row = lane & 15;
  const int g   = lane >> 4;

  const float* xb = X + ((size_t)b * SS + m0) * DD;

  v8f acc[10];
  #pragma unroll
  for (int n = 0; n < 10; ++n) acc[n] = (v8f)(0.f);

  #pragma unroll
  for (int ks = 0; ks < 5; ++ks){
    const int k0 = ks * 32;
    Frag a;
    const float* rp = xb + (size_t)row * DD + k0 + g * 8;
    float4 f0 = *(const float4*)(rp);
    float4 f1 = *(const float4*)(rp + 4);
    float4 f2 = *(const float4*)(rp + 16);
    float4 f3 = *(const float4*)(rp + 20);
    a.u[0] = pk_bf16(f0.x, f0.y); a.u[1] = pk_bf16(f0.z, f0.w);
    a.u[2] = pk_bf16(f1.x, f1.y); a.u[3] = pk_bf16(f1.z, f1.w);
    a.u[4] = pk_bf16(f2.x, f2.y); a.u[5] = pk_bf16(f2.z, f2.w);
    a.u[6] = pk_bf16(f3.x, f3.y); a.u[7] = pk_bf16(f3.z, f3.w);
    #pragma unroll
    for (int np = 0; np < 5; ++np){          // process n-tiles in pairs
      Frag b0, b1;
      const uint16_t* cp0 = Wb + (size_t)((2*np  ) * 16 + row) * DD + k0 + g * 8;
      const uint16_t* cp1 = Wb + (size_t)((2*np+1) * 16 + row) * DD + k0 + g * 8;
      b0.q4[0] = *(const uint4*)(cp0);
      b0.q4[1] = *(const uint4*)(cp0 + 16);
      b1.q4[0] = *(const uint4*)(cp1);
      b1.q4[1] = *(const uint4*)(cp1 + 16);
      acc[2*np  ] = __builtin_amdgcn_wmma_f32_16x16x32_bf16(
          false, a.v, false, b0.v, (short)0, acc[2*np  ], false, false);
      acc[2*np+1] = __builtin_amdgcn_wmma_f32_16x16x32_bf16(
          false, a.v, false, b1.v, (short)0, acc[2*np+1], false, false);
    }
  }

  #pragma unroll
  for (int nt = 0; nt < 10; ++nt){
    const int col = nt * 16 + row;
    const float bv = bias[col];
    if (!TRANS){
      uint16_t* op = out + ((size_t)b * SS + m0) * DD + col;
      #pragma unroll
      for (int j = 0; j < 8; ++j)
        op[(size_t)(j + 8 * g) * DD] = bf16_1(acc[nt][j] + bv);
    } else {
      uint32_t pk0 = pk_bf16(acc[nt][0] + bv, acc[nt][1] + bv);
      uint32_t pk1 = pk_bf16(acc[nt][2] + bv, acc[nt][3] + bv);
      uint32_t pk2 = pk_bf16(acc[nt][4] + bv, acc[nt][5] + bv);
      uint32_t pk3 = pk_bf16(acc[nt][6] + bv, acc[nt][7] + bv);
      uint16_t* op = out + ((size_t)b * DD + col) * SS + m0 + 8 * g;
      *(uint4*)op = make_uint4(pk0, pk1, pk2, pk3);
    }
  }
}

// ---- flash attention: per-wave 16 q-rows, SY streamed in 64-col chunks ----
__global__ void __launch_bounds__(256, 1)
attn_kernel(const uint16_t* __restrict__ Q,   // [B,SX,D] bf16
            const uint16_t* __restrict__ K,   // [B,SY,D] bf16
            const uint16_t* __restrict__ Vt,  // [B,D,SY] bf16
            const float* __restrict__ X,      // residual [B,SX,D]
            float* __restrict__ Out)          // [B,SX,D]
{
  __shared__ uint16_t ldsP[8 * 16 * 64];   // 16KB: per-wave P transpose buffer
  const int lane = threadIdx.x & 31;
  const int wave = threadIdx.x >> 5;
  const int tile = blockIdx.x * 8 + wave;
  const int tilesPerB = SS / 16;
  if (tile >= BB * tilesPerB) return;
  const int b   = tile / tilesPerB;
  const int m0  = (tile % tilesPerB) * 16;
  const int row = lane & 15;
  const int g   = lane >> 4;
  uint16_t* myP = ldsP + wave * 16 * 64;

  // Q fragments: resident for the whole wave
  Frag qf[5];
  {
    const uint16_t* qrow = Q + ((size_t)b * SS + m0 + row) * DD;
    #pragma unroll
    for (int ks = 0; ks < 5; ++ks){
      const uint16_t* p = qrow + ks * 32 + g * 8;
      qf[ks].q4[0] = *(const uint4*)(p);
      qf[ks].q4[1] = *(const uint4*)(p + 16);
    }
  }

  float ms[8], ls[8];
  #pragma unroll
  for (int j = 0; j < 8; ++j){ ms[j] = -3.0e38f; ls[j] = 0.f; }
  v8f o[10];
  #pragma unroll
  for (int n = 0; n < 10; ++n) o[n] = (v8f)(0.f);

  const uint16_t* kb = K  + (size_t)b * SS * DD;
  const uint16_t* vb = Vt + (size_t)b * DD * SS;

  for (int sy0 = 0; sy0 < SS; sy0 += 64){
    // prefetch next chunk's K/V into caches
    if (sy0 + 64 < SS){
      __builtin_prefetch(kb + (size_t)(sy0 + 64 + (lane << 1)) * DD, 0, 3);
      __builtin_prefetch(vb + (size_t)(lane * 5) * SS + sy0 + 64, 0, 3);
    }
    // S = Q * K^T  (16x64 tile; K rows are contiguous B-fragment columns)
    v8f s[4];
    #pragma unroll
    for (int c = 0; c < 4; ++c) s[c] = (v8f)(0.f);
    #pragma unroll
    for (int ks = 0; ks < 5; ++ks){
      Frag kf[4];
      #pragma unroll
      for (int c = 0; c < 4; ++c){
        const uint16_t* p = kb + (size_t)(sy0 + c * 16 + row) * DD + ks * 32 + g * 8;
        kf[c].q4[0] = *(const uint4*)(p);
        kf[c].q4[1] = *(const uint4*)(p + 16);
      }
      #pragma unroll
      for (int c = 0; c < 4; ++c)
        s[c] = __builtin_amdgcn_wmma_f32_16x16x32_bf16(
            false, qf[ks].v, false, kf[c].v, (short)0, s[c], false, false);
    }
    // online softmax (rows live across 16 lanes of each half-wave)
    #pragma unroll
    for (int j = 0; j < 8; ++j){
      float mx = fmaxf(fmaxf(s[0][j], s[1][j]), fmaxf(s[2][j], s[3][j]));
      #pragma unroll
      for (int off = 1; off < 16; off <<= 1)
        mx = fmaxf(mx, __shfl_xor(mx, off, 16));
      float mnew  = fmaxf(ms[j], mx);
      float scale = __expf(ms[j] - mnew);
      ms[j] = mnew;
      float rs = 0.f;
      #pragma unroll
      for (int c = 0; c < 4; ++c){
        float p = __expf(s[c][j] - mnew);
        s[c][j] = p;
        rs += p;
      }
      #pragma unroll
      for (int off = 1; off < 16; off <<= 1)
        rs += __shfl_xor(rs, off, 16);
      ls[j] = ls[j] * scale + rs;
      #pragma unroll
      for (int n = 0; n < 10; ++n) o[n][j] *= scale;
    }
    // C-layout -> A-layout via per-wave LDS (row-major 16x64 bf16)
    #pragma unroll
    for (int c = 0; c < 4; ++c){
      #pragma unroll
      for (int j = 0; j < 8; ++j)
        myP[(size_t)(j + 8 * g) * 64 + c * 16 + row] = bf16_1(s[c][j]);
    }
    asm volatile("s_wait_dscnt 0" ::: "memory");
    Frag pa[2];
    #pragma unroll
    for (int kk = 0; kk < 2; ++kk){
      const uint16_t* p = myP + row * 64 + kk * 32 + g * 8;
      pa[kk].q4[0] = *(const uint4*)(p);
      pa[kk].q4[1] = *(const uint4*)(p + 16);
    }
    // O += P * V   (V^T layout makes B-fragment columns contiguous)
    #pragma unroll
    for (int nt = 0; nt < 10; ++nt){
      Frag vf0, vf1;
      const uint16_t* p0 = vb + (size_t)(nt * 16 + row) * SS + sy0;
      vf0.q4[0] = *(const uint4*)(p0 + g * 8);
      vf0.q4[1] = *(const uint4*)(p0 + g * 8 + 16);
      vf1.q4[0] = *(const uint4*)(p0 + 32 + g * 8);
      vf1.q4[1] = *(const uint4*)(p0 + 32 + g * 8 + 16);
      o[nt] = __builtin_amdgcn_wmma_f32_16x16x32_bf16(
          false, pa[0].v, false, vf0.v, (short)0, o[nt], false, false);
      o[nt] = __builtin_amdgcn_wmma_f32_16x16x32_bf16(
          false, pa[1].v, false, vf1.v, (short)0, o[nt], false, false);
    }
  }
  // epilogue: normalize + residual
  float inv[8];
  #pragma unroll
  for (int j = 0; j < 8; ++j) inv[j] = __builtin_amdgcn_rcpf(ls[j]);
  const float* xr  = X   + ((size_t)b * SS + m0) * DD;
  float* orow      = Out + ((size_t)b * SS + m0) * DD;
  #pragma unroll
  for (int nt = 0; nt < 10; ++nt){
    const int col = nt * 16 + row;
    #pragma unroll
    for (int j = 0; j < 8; ++j){
      const size_t idx = (size_t)(j + 8 * g) * DD + col;
      orow[idx] = o[nt][j] * inv[j] + xr[idx];
    }
  }
}

extern "C" void kernel_launch(void* const* d_in, const int* in_sizes, int n_in,
                              void* d_out, int out_size, void* d_ws, size_t ws_size,
                              hipStream_t stream)
{
  (void)in_sizes; (void)n_in; (void)out_size; (void)ws_size;
  const float* x  = (const float*)d_in[0];
  const float* y  = (const float*)d_in[1];
  const float* Wq = (const float*)d_in[2];
  const float* bq = (const float*)d_in[3];
  const float* Wk = (const float*)d_in[4];
  const float* bk = (const float*)d_in[5];
  const float* Wv = (const float*)d_in[6];
  const float* bv = (const float*)d_in[7];
  float* out = (float*)d_out;

  uint8_t* ws = (uint8_t*)d_ws;
  uint16_t* wqb = (uint16_t*)ws;
  uint16_t* wkb = wqb + DD * DD;
  uint16_t* wvb = wkb + DD * DD;
  size_t woff = ((size_t)3 * DD * DD * 2 + 255) & ~(size_t)255;
  uint16_t* qb  = (uint16_t*)(ws + woff);
  uint16_t* kbf = qb  + (size_t)BB * SS * DD;
  uint16_t* vtb = kbf + (size_t)BB * SS * DD;

  {
    int n = 3 * DD * DD;
    wcvt_kernel<<<(n + 255) / 256, 256, 0, stream>>>(Wq, Wk, Wv, wqb);
  }
  const int tiles  = BB * (SS / 16);
  const int blocks = (tiles + 7) / 8;            // 8 waves (256 threads) per block
  proj_kernel<false><<<blocks, 256, 0, stream>>>(x, wqb, bq, qb);
  proj_kernel<false><<<blocks, 256, 0, stream>>>(y, wkb, bk, kbf);
  proj_kernel<true ><<<blocks, 256, 0, stream>>>(y, wvb, bv, vtb);
  attn_kernel<<<blocks, 256, 0, stream>>>(qb, kbf, vtb, x, out);
}
